// GraphConv_54511724921066
// MI455X (gfx1250) — matrix operand order repaired and checked by reading, not compiled, not run on hardware
//
#include <hip/hip_runtime.h>
#include <hip/hip_bf16.h>

typedef __attribute__((ext_vector_type(16))) __bf16 v16bf;
typedef __attribute__((ext_vector_type(8)))  float  v8f;

#define B_    4
#define N_    4096
#define K_    20
#define C_    64
#define KIN   160          // 132 padded to 5*32
#define KREAL 132
#define MTOT  (B_*N_*K_)   // 327680

union AFrag {
    v16bf v;
    unsigned int w[8];
    __bf16 h[16];
};

union Pack2 { __bf16 h[2]; unsigned int u; };

// ---------------- 0: zero stats ----------------
__global__ void zero_stats_kernel(float* p) {
    p[threadIdx.x] = 0.0f;            // 256 threads: stats1(128)+stats2(128) contiguous
}

// ---------------- 1: transpose x (B,C,N) -> xT (B,N,C) ----------------
__global__ __launch_bounds__(256) void transpose_x_kernel(const float* __restrict__ x,
                                                          float* __restrict__ xT) {
    int t = blockIdx.x * 256 + threadIdx.x;       // over B*N*64
    int c = t & 63;
    int n = (t >> 6) & (N_ - 1);
    int b = t >> 18;
    xT[t] = x[((long)(b * C_ + c)) * N_ + n];
}

// ---------------- 2: kNN top-20 ----------------
__global__ __launch_bounds__(256) void knn_kernel(const float* __restrict__ pos,
                                                  int* __restrict__ idx) {
    const int b = blockIdx.y;
    const int n = blockIdx.x * 256 + threadIdx.x;
    __shared__ float sx[256], sy[256], sz[256], sq[256];
    const float* px = pos + (long)b * 3 * N_;
    const float qx = px[n], qy = px[N_ + n], qz = px[2 * N_ + n];
    const float qsq = qx * qx + qy * qy + qz * qz;

    float d[K_]; int id[K_];
#pragma unroll
    for (int t = 0; t < K_; ++t) { d[t] = 3.4e38f; id[t] = 0; }

    for (int tile = 0; tile < N_; tile += 256) {
        const int j = tile + threadIdx.x;
        const float cx = px[j], cy = px[N_ + j], cz = px[2 * N_ + j];
        __syncthreads();
        sx[threadIdx.x] = cx; sy[threadIdx.x] = cy; sz[threadIdx.x] = cz;
        sq[threadIdx.x] = cx * cx + cy * cy + cz * cz;
        __syncthreads();
        for (int jj = 0; jj < 256; ++jj) {
            const float d2 = qsq + sq[jj] - 2.0f * (qx * sx[jj] + qy * sy[jj] + qz * sz[jj]);
            if (d2 < d[K_ - 1]) {
                d[K_ - 1] = d2; id[K_ - 1] = tile + jj;
#pragma unroll
                for (int t = K_ - 1; t > 0; --t) {
                    if (d[t] < d[t - 1]) {
                        float td = d[t]; d[t] = d[t - 1]; d[t - 1] = td;
                        int ti = id[t]; id[t] = id[t - 1]; id[t - 1] = ti;
                    }
                }
            }
        }
    }
    int* op = idx + ((long)b * N_ + n) * K_;
#pragma unroll
    for (int t = 0; t < K_; ++t) op[t] = id[t];
}

// ---------------- 3: edge features (bf16, K padded to 160) ----------------
__global__ __launch_bounds__(256) void edge_kernel(const float* __restrict__ pos,
                                                   const float* __restrict__ xT,
                                                   const int* __restrict__ idx,
                                                   __bf16* __restrict__ E) {
    const long row = (long)blockIdx.x * 4 + (threadIdx.x >> 6);   // edge row
    const int c = threadIdx.x & 63;
    const int kk = (int)(row % K_);
    const long bn = row / K_;                 // b*N + n
    const int n = (int)(bn & (N_ - 1));
    const int b = (int)(bn >> 12);
    const int j = idx[row];

    const float xn = xT[bn * 64 + c];
    const float xj = xT[((long)b * N_ + j) * 64 + c];
    __bf16* er = E + row * KIN;
    er[c]      = (__bf16)(xj - xn);           // feat - x_rep
    er[68 + c] = (__bf16)xn;                  // x_rep broadcast

    if (c < 3) {
        const float* pp = pos + (long)b * 3 * N_;
        er[64 + c] = (__bf16)(pp[c * N_ + j] - pp[c * N_ + n]);   // fpos - p_rep
    } else if (c == 3) {
        const float* pp = pos + (long)b * 3 * N_;
        const float dx = pp[n] - pp[j] + 1e-6f;
        const float dy = pp[N_ + n] - pp[N_ + j] + 1e-6f;
        const float dz = pp[2 * N_ + n] - pp[2 * N_ + j] + 1e-6f;
        const float dn = sqrtf(dx * dx + dy * dy + dz * dz);
        // reference's transpose-scramble: value (n,kk) lands at flat i = kk*N + n
        const long dest = (long)b * (N_ * K_) + (long)kk * N_ + n;
        E[dest * KIN + 67] = (__bf16)dn;
    } else if (c < 32) {
        er[132 + (c - 4)] = (__bf16)0.0f;     // zero pad K 132..159
    }
}

// ---------------- 4: GEMM1  E(M,160)bf16 x w1^T -> y1(M,64)f32 + stats ----------------
// LDS weights pre-swizzled fragment-ready: Wl[((kt*4+nt)*32 + lane)*8 + v] = bf16 pair
// Each wave computes 32 rows (2 M-tiles) so each B fragment feeds 2 WMMAs.
__global__ __launch_bounds__(256) void gemm1_kernel(const __bf16* __restrict__ E,
                                                    const float* __restrict__ w1,  // (64,132)
                                                    float* __restrict__ y1,
                                                    float* __restrict__ stats) {
    __shared__ unsigned int Wl[5 * 4 * 32 * 8];   // 20 KB
    __shared__ float lsum[64], lsq[64];
    const int tid = threadIdx.x;
    for (int i = tid; i < 5 * 4 * 32 * 8; i += 256) {
        const int v  = i & 7;
        const int ln = (i >> 3) & 31;
        const int nt = (i >> 8) & 3;
        const int kt = i >> 10;
        const int kk0 = kt * 32 + (ln >> 4) * 8 + (v >> 2) * 16 + 2 * (v & 3);
        const int o   = nt * 16 + (ln & 15);
        Pack2 p;
        p.h[0] = (kk0     < KREAL) ? (__bf16)w1[o * KREAL + kk0]     : (__bf16)0.0f;
        p.h[1] = (kk0 + 1 < KREAL) ? (__bf16)w1[o * KREAL + kk0 + 1] : (__bf16)0.0f;
        Wl[i] = p.u;
    }
    if (tid < 64) { lsum[tid] = 0.0f; lsq[tid] = 0.0f; }
    __syncthreads();

    const int wave = tid >> 5;
    const int lane = tid & 31;
    const int hl = lane >> 4;
    const int l16 = lane & 15;
    const long m0 = (long)blockIdx.x * 256 + wave * 32;   // 2 M-tiles per wave

    v8f acc0[4] = {}, acc1[4] = {};
#pragma unroll
    for (int kt = 0; kt < 5; ++kt) {
        AFrag a0, a1;
        const uint4* pA0 = (const uint4*)(E + (m0 + l16) * KIN + kt * 32 + hl * 8);
        const uint4* pA1 = (const uint4*)(E + (m0 + 16 + l16) * KIN + kt * 32 + hl * 8);
        ((uint4*)a0.w)[0] = pA0[0];   // K = base+0..7
        ((uint4*)a0.w)[1] = pA0[2];   // K = base+16..23
        ((uint4*)a1.w)[0] = pA1[0];
        ((uint4*)a1.w)[1] = pA1[2];
#pragma unroll
        for (int nt = 0; nt < 4; ++nt) {
            AFrag b;
            const uint4* pB = (const uint4*)(Wl + (((kt * 4 + nt) * 32 + lane) << 3));
            ((uint4*)b.w)[0] = pB[0];
            ((uint4*)b.w)[1] = pB[1];
            acc0[nt] = __builtin_amdgcn_wmma_f32_16x16x32_bf16(
                false, a0.v, false, b.v, (short)0, acc0[nt], false, false);
            acc1[nt] = __builtin_amdgcn_wmma_f32_16x16x32_bf16(
                false, a1.v, false, b.v, (short)0, acc1[nt], false, false);
        }
    }
#pragma unroll
    for (int nt = 0; nt < 4; ++nt) {
        float ps = 0.0f, pq = 0.0f;
#pragma unroll
        for (int r = 0; r < 8; ++r) {
            const float v0 = acc0[nt][r];
            const float v1 = acc1[nt][r];
            y1[(m0 + r + hl * 8) * 64 + nt * 16 + l16] = v0;
            y1[(m0 + 16 + r + hl * 8) * 64 + nt * 16 + l16] = v1;
            ps += v0 + v1; pq += v0 * v0 + v1 * v1;
        }
        atomicAdd(&lsum[nt * 16 + l16], ps);
        atomicAdd(&lsq[nt * 16 + l16], pq);
    }
    __syncthreads();
    if (tid < 64) {
        atomicAdd(&stats[tid], lsum[tid]);
        atomicAdd(&stats[64 + tid], lsq[tid]);
    }
}

// ---------------- 5: BN finalize (conv bias cancels inside BN) ----------------
__global__ void bn_finalize_kernel(const float* __restrict__ stats,
                                   const float* __restrict__ gamma,
                                   const float* __restrict__ beta,
                                   float* __restrict__ bn) {
    const int c = threadIdx.x;                // 64 threads
    const float cnt = (float)MTOT;
    const float mu = stats[c] / cnt;
    const float var = stats[64 + c] / cnt - mu * mu;
    const float s = gamma[c] * rsqrtf(var + 1e-5f);
    bn[c] = s;
    bn[64 + c] = beta[c] - mu * s;
}

// ---------------- 6: GEMM2  BN+lrelu(y1) x w2^T -> y2 + stats ----------------
__global__ __launch_bounds__(256) void gemm2_kernel(const float* __restrict__ y1,
                                                    const float* __restrict__ w2,  // (64,64)
                                                    const float* __restrict__ bn1, // s[64],t[64]
                                                    float* __restrict__ y2,
                                                    float* __restrict__ stats) {
    __shared__ unsigned int Wl[2 * 4 * 32 * 8];   // 8 KB, fragment-ready
    __shared__ float sS[64], sT[64];
    __shared__ float lsum[64], lsq[64];
    const int tid = threadIdx.x;
    for (int i = tid; i < 2 * 4 * 32 * 8; i += 256) {
        const int v  = i & 7;
        const int ln = (i >> 3) & 31;
        const int nt = (i >> 8) & 3;
        const int kt = i >> 10;
        const int kk0 = kt * 32 + (ln >> 4) * 8 + (v >> 2) * 16 + 2 * (v & 3);
        const int o   = nt * 16 + (ln & 15);
        Pack2 p;
        p.h[0] = (__bf16)w2[o * 64 + kk0];
        p.h[1] = (__bf16)w2[o * 64 + kk0 + 1];
        Wl[i] = p.u;
    }
    if (tid < 64) {
        sS[tid] = bn1[tid]; sT[tid] = bn1[64 + tid];
        lsum[tid] = 0.0f; lsq[tid] = 0.0f;
    }
    __syncthreads();

    const int wave = tid >> 5;
    const int lane = tid & 31;
    const int hl = lane >> 4;
    const int l16 = lane & 15;
    const long m0 = (long)blockIdx.x * 256 + wave * 32;   // 2 M-tiles per wave

    v8f acc0[4] = {}, acc1[4] = {};
#pragma unroll
    for (int kt = 0; kt < 2; ++kt) {
        const int kb = kt * 32 + hl * 8;
        AFrag a0, a1;
#pragma unroll
        for (int mt = 0; mt < 2; ++mt) {
            const float4* pA = (const float4*)(y1 + (m0 + mt * 16 + l16) * 64 + kb);
            float4 f0 = pA[0];            // K = kb+0..3
            float4 f1 = pA[1];            // K = kb+4..7
            float4 f2 = pA[4];            // K = kb+16..19
            float4 f3 = pA[5];            // K = kb+20..23
            float e0[8] = { f0.x, f0.y, f0.z, f0.w, f1.x, f1.y, f1.z, f1.w };
            float e1[8] = { f2.x, f2.y, f2.z, f2.w, f3.x, f3.y, f3.z, f3.w };
            AFrag& a = mt ? a1 : a0;
#pragma unroll
            for (int e = 0; e < 8; ++e) {
                float h0 = e0[e] * sS[kb + e] + sT[kb + e];
                float h1 = e1[e] * sS[kb + 16 + e] + sT[kb + 16 + e];
                h0 = h0 >= 0.0f ? h0 : 0.2f * h0;
                h1 = h1 >= 0.0f ? h1 : 0.2f * h1;
                a.h[e]     = (__bf16)h0;
                a.h[8 + e] = (__bf16)h1;
            }
        }
#pragma unroll
        for (int nt = 0; nt < 4; ++nt) {
            AFrag b;
            const uint4* pB = (const uint4*)(Wl + (((kt * 4 + nt) * 32 + lane) << 3));
            ((uint4*)b.w)[0] = pB[0];
            ((uint4*)b.w)[1] = pB[1];
            acc0[nt] = __builtin_amdgcn_wmma_f32_16x16x32_bf16(
                false, a0.v, false, b.v, (short)0, acc0[nt], false, false);
            acc1[nt] = __builtin_amdgcn_wmma_f32_16x16x32_bf16(
                false, a1.v, false, b.v, (short)0, acc1[nt], false, false);
        }
    }
#pragma unroll
    for (int nt = 0; nt < 4; ++nt) {
        float ps = 0.0f, pq = 0.0f;
#pragma unroll
        for (int r = 0; r < 8; ++r) {
            const float v0 = acc0[nt][r];
            const float v1 = acc1[nt][r];
            y2[(m0 + r + hl * 8) * 64 + nt * 16 + l16] = v0;
            y2[(m0 + 16 + r + hl * 8) * 64 + nt * 16 + l16] = v1;
            ps += v0 + v1; pq += v0 * v0 + v1 * v1;
        }
        atomicAdd(&lsum[nt * 16 + l16], ps);
        atomicAdd(&lsq[nt * 16 + l16], pq);
    }
    __syncthreads();
    if (tid < 64) {
        atomicAdd(&stats[tid], lsum[tid]);
        atomicAdd(&stats[64 + tid], lsq[tid]);
    }
}

// ---------------- 7: BN2 + lrelu + max over k -> out (B,64,N) ----------------
__global__ __launch_bounds__(256) void maxpool_kernel(const float* __restrict__ y2,
                                                      const float* __restrict__ bn2,
                                                      float* __restrict__ out) {
    const int t = blockIdx.x * 256 + threadIdx.x;   // over B*N*64
    const int o = t & 63;
    const int n = (t >> 6) & (N_ - 1);
    const int b = t >> 18;
    const float s = bn2[o], tt = bn2[64 + o];
    const float* p = y2 + ((long)(b * N_ + n) * K_) * 64 + o;
    float m = -3.4e38f;
#pragma unroll
    for (int kk = 0; kk < K_; ++kk) {
        float v = p[kk * 64] * s + tt;
        v = v >= 0.0f ? v : 0.2f * v;
        m = fmaxf(m, v);
    }
    out[((long)(b * 64 + o)) * N_ + n] = m;
}

// ---------------- launch ----------------
extern "C" void kernel_launch(void* const* d_in, const int* in_sizes, int n_in,
                              void* d_out, int out_size, void* d_ws, size_t ws_size,
                              hipStream_t stream) {
    const float* x   = (const float*)d_in[0];
    const float* pos = (const float*)d_in[1];
    const float* w1  = (const float*)d_in[2];
    // d_in[3] = b1 : cancels inside BN
    const float* g1  = (const float*)d_in[4];
    const float* be1 = (const float*)d_in[5];
    const float* w2  = (const float*)d_in[6];
    // d_in[7] = b2 : cancels inside BN
    const float* g2  = (const float*)d_in[8];
    const float* be2 = (const float*)d_in[9];
    float* out = (float*)d_out;

    char* ws = (char*)d_ws;
    float* stats1 = (float*)(ws + 0);            // 128 f32
    float* stats2 = (float*)(ws + 512);          // 128 f32
    float* bn1    = (float*)(ws + 1024);         // s,t
    float* bn2    = (float*)(ws + 1536);
    int*   idx    = (int*)(ws + 4096);                       // 1,310,720 B
    float* xT     = (float*)(ws + 4096 + 1310720);           // 4,194,304 B
    __bf16* E     = (__bf16*)(ws + 4096 + 1310720 + 4194304);            // 104,857,600 B
    float* y1     = (float*)(ws + 4096 + 1310720 + 4194304 + 104857600); // 83,886,080 B
    float* y2     = (float*)E;                   // alias: E dead after GEMM1

    zero_stats_kernel<<<1, 256, 0, stream>>>(stats1);        // stats1+stats2 contiguous
    transpose_x_kernel<<<(B_ * N_ * 64) / 256, 256, 0, stream>>>(x, xT);
    knn_kernel<<<dim3(N_ / 256, B_), 256, 0, stream>>>(pos, idx);
    edge_kernel<<<MTOT / 4, 256, 0, stream>>>(pos, xT, idx, E);
    gemm1_kernel<<<MTOT / 256, 256, 0, stream>>>(E, w1, y1, stats1);
    bn_finalize_kernel<<<1, 64, 0, stream>>>(stats1, g1, be1, bn1);
    gemm2_kernel<<<MTOT / 256, 256, 0, stream>>>(y1, w2, bn1, y2, stats2);
    bn_finalize_kernel<<<1, 64, 0, stream>>>(stats2, g2, be2, bn2);
    maxpool_kernel<<<(B_ * N_ * 64) / 256, 256, 0, stream>>>(y2, bn2, out);
}
